// MultiGAT_20985210208444
// MI455X (gfx1250) — matrix operand order repaired and checked by reading, not compiled, not run on hardware
//
#include <hip/hip_runtime.h>
#include <hip/hip_bf16.h>
#include <stdint.h>

typedef __attribute__((ext_vector_type(16))) __bf16 v16bf;
typedef __attribute__((ext_vector_type(8)))  float  v8f;

#define FDIM 256   // H*C
#define NHEAD 4
#define CPH 64

// ---------------- helpers ----------------
__device__ __forceinline__ float lrelu(float v) { return v > 0.f ? v : 0.2f * v; }

// monotone float<->uint key for atomicMax on floats
__device__ __forceinline__ unsigned fkey(float f) {
    unsigned b = __float_as_uint(f);
    return (b & 0x80000000u) ? ~b : (b | 0x80000000u);
}
__device__ __forceinline__ float funkey(unsigned k) {
    unsigned b = (k & 0x80000000u) ? (k & 0x7fffffffu) : ~k;
    return __uint_as_float(b);
}

// ---------------- prep: f32 -> bf16 ----------------
__global__ __launch_bounds__(256) void cvt_bf16_k(const float* __restrict__ src,
                                                  __bf16* __restrict__ dst, size_t n) {
    size_t i = (size_t)blockIdx.x * 256 + threadIdx.x;
    if (i < n) dst[i] = (__bf16)src[i];
}

// W[k][n] (f32, 256x256) -> Wt[n][k] (bf16) so async staging is layout-preserving
__global__ __launch_bounds__(256) void cvt_w_transpose_k(const float* __restrict__ W,
                                                         __bf16* __restrict__ Wt) {
    int id = blockIdx.x * 256 + threadIdx.x;     // 65536 threads
    int k = id >> 8, n = id & 255;
    Wt[n * FDIM + k] = (__bf16)W[id];            // W[id] = W[k*256+n]
}

// ---------------- GEMM: H = Xb @ W  (WMMA bf16 -> f32) ----------------
// block = 256 thr = 8 waves; block computes 128 rows x 256 cols.
// wave w -> rows [m0+16w, m0+16w+16), 16 column tiles of 16.
__global__ __launch_bounds__(256) void gemm_wmma_k(const __bf16* __restrict__ Xb,
                                                   const __bf16* __restrict__ Wt, // [n][k]
                                                   float* __restrict__ Hout, int M) {
    extern __shared__ char smem[];
    __bf16* sX = (__bf16*)smem;                      // 128 x 256  (64 KB)
    __bf16* sW = (__bf16*)(smem + 128 * FDIM * 2);   // 256 x 256  (128 KB), [n][k]
    const int tid = threadIdx.x;
    const int m0 = blockIdx.x * 128;
    const uint32_t ldsX = (uint32_t)(uintptr_t)sX;
    const uint32_t ldsW = (uint32_t)(uintptr_t)sW;

    // async stage X tile: 128 rows x 32 chunks(16B), rows clamped to M-1
    for (int i = tid; i < 128 * 32; i += 256) {
        int r = i >> 5, ck = i & 31;
        int gr = m0 + r; if (gr >= M) gr = M - 1;
        const char* g = (const char*)(Xb + (size_t)gr * FDIM) + ck * 16;
        uint32_t l = ldsX + (uint32_t)(r * 512 + ck * 16);
        asm volatile("global_load_async_to_lds_b128 %0, %1, off" :: "v"(l), "v"(g) : "memory");
    }
    // async stage full Wt: 8192 chunks of 16B
    for (int i = tid; i < 8192; i += 256) {
        const char* g = (const char*)Wt + (size_t)i * 16;
        uint32_t l = ldsW + (uint32_t)(i * 16);
        asm volatile("global_load_async_to_lds_b128 %0, %1, off" :: "v"(l), "v"(g) : "memory");
    }
    asm volatile("s_wait_asynccnt 0" ::: "memory");
    __syncthreads();

    const int wave = tid >> 5, lane = tid & 31;
    const int half = lane >> 4, l16 = lane & 15;

    v8f acc[16];
#pragma unroll
    for (int ct = 0; ct < 16; ++ct)
#pragma unroll
        for (int r = 0; r < 8; ++r) acc[ct][r] = 0.f;

    const __bf16* aRow = sX + (wave * 16 + l16) * FDIM;
    for (int kb = 0; kb < FDIM; kb += 32) {
        // A-frag (16x32 bf16): lane half 0: K=kb+{0..7,16..23}; half 1: K=kb+{8..15,24..31}
        v16bf a;
#pragma unroll
        for (int e = 0; e < 8; ++e) {
            a[e]     = aRow[kb + half * 8 + e];
            a[e + 8] = aRow[kb + 16 + half * 8 + e];
        }
#pragma unroll
        for (int ct = 0; ct < 16; ++ct) {
            // B-frag (32x16 bf16): lane = col l16; half 0: K=kb+{0..7,16..23}; half 1: K=kb+{8..15,24..31}
            const __bf16* bCol = sW + (ct * 16 + l16) * FDIM;
            v16bf b;
#pragma unroll
            for (int e = 0; e < 8; ++e) {
                b[e]     = bCol[kb + half * 8 + e];
                b[e + 8] = bCol[kb + 16 + half * 8 + e];
            }
            acc[ct] = __builtin_amdgcn_wmma_f32_16x16x32_bf16(
                false, a, false, b, (short)0, acc[ct], false, false);
        }
    }
    // D layout: VGPR r -> M = r + 8*half, N = l16
#pragma unroll
    for (int ct = 0; ct < 16; ++ct) {
        int col = ct * 16 + l16;
#pragma unroll
        for (int r = 0; r < 8; ++r) {
            int gm = m0 + wave * 16 + half * 8 + r;
            if (gm < M) Hout[(size_t)gm * FDIM + col] = acc[ct][r];
        }
    }
}

// ---------------- attention scores: e_src/e_dst [N,H] ----------------
__global__ __launch_bounds__(256) void att_k(const float* __restrict__ h,
                                             const float* __restrict__ a_s,
                                             const float* __restrict__ a_d,
                                             float* __restrict__ e_s,
                                             float* __restrict__ e_d, int N) {
    int id = blockIdx.x * 256 + threadIdx.x;
    if (id >= N * NHEAD) return;
    int n = id >> 2, hd = id & 3;
    const float4* hp  = (const float4*)(h + (size_t)n * FDIM + hd * CPH);
    const float4* sp  = (const float4*)(a_s + hd * CPH);
    const float4* dp  = (const float4*)(a_d + hd * CPH);
    float ss = 0.f, sd = 0.f;
#pragma unroll
    for (int i = 0; i < 16; ++i) {
        float4 hv = hp[i], av = sp[i], dv = dp[i];
        ss += hv.x * av.x + hv.y * av.y + hv.z * av.z + hv.w * av.w;
        sd += hv.x * dv.x + hv.y * dv.y + hv.z * dv.z + hv.w * dv.w;
    }
    e_s[id] = ss; e_d[id] = sd;
}

// init segment max with self-loop value (every node has one); zero denom
__global__ __launch_bounds__(256) void init_max_k(const float* __restrict__ e_s,
                                                  const float* __restrict__ e_d,
                                                  unsigned* __restrict__ mato,
                                                  float* __restrict__ denom, int N) {
    int id = blockIdx.x * 256 + threadIdx.x;
    if (id >= N * NHEAD) return;
    mato[id]  = fkey(lrelu(e_s[id] + e_d[id]));
    denom[id] = 0.f;
}

__global__ __launch_bounds__(256) void edge_max_k(const int* __restrict__ srcI,
                                                  const int* __restrict__ dstI,
                                                  const float* __restrict__ e_s,
                                                  const float* __restrict__ e_d,
                                                  unsigned* __restrict__ mato, int E) {
    int id = blockIdx.x * 256 + threadIdx.x;
    if (id >= E * NHEAD) return;
    int e = id >> 2, hd = id & 3;
    int s = srcI[e], d = dstI[e];
    float v = lrelu(e_s[s * NHEAD + hd] + e_d[d * NHEAD + hd]);
    atomicMax(&mato[d * NHEAD + hd], fkey(v));
}

// ex + denom over edges and self-loops (ids [0, (E+N)*H))
__global__ __launch_bounds__(256) void ex_denom_k(const int* __restrict__ srcI,
                                                  const int* __restrict__ dstI,
                                                  const float* __restrict__ e_s,
                                                  const float* __restrict__ e_d,
                                                  const unsigned* __restrict__ mato,
                                                  float* __restrict__ exbuf,
                                                  float* __restrict__ denom,
                                                  int E, int N) {
    size_t id = (size_t)blockIdx.x * 256 + threadIdx.x;
    if (id >= (size_t)(E + N) * NHEAD) return;
    int e = (int)(id >> 2), hd = (int)(id & 3);
    int s, d;
    if (e < E) { s = srcI[e]; d = dstI[e]; } else { s = d = e - E; }
    float v  = lrelu(e_s[s * NHEAD + hd] + e_d[d * NHEAD + hd]);
    float ex = __expf(v - funkey(mato[d * NHEAD + hd]));
    exbuf[id] = ex;
    atomicAdd(&denom[d * NHEAD + hd], ex);
}

// initialize agg with the self-loop contribution (plain store: 1 self-loop per node)
__global__ __launch_bounds__(256) void agg_self_k(const float* __restrict__ h,
                                                  const float* __restrict__ exbuf,
                                                  const float* __restrict__ denom,
                                                  float* __restrict__ agg, int N, int E) {
    size_t i = (size_t)blockIdx.x * 256 + threadIdx.x;
    if (i >= (size_t)N * FDIM) return;
    int n = (int)(i >> 8), c = (int)(i & 255), hd = c >> 6;
    float alpha = exbuf[((size_t)E + n) * NHEAD + hd] / (denom[n * NHEAD + hd] + 1e-16f);
    agg[i] = h[i] * alpha;
}

// one wave32 per edge: lane covers 8 channels (head-aligned), f32 atomics into agg
__global__ __launch_bounds__(256) void agg_edge_k(const float* __restrict__ h,
                                                  const int* __restrict__ srcI,
                                                  const int* __restrict__ dstI,
                                                  const float* __restrict__ exbuf,
                                                  const float* __restrict__ denom,
                                                  float* __restrict__ agg, int E) {
    int gid  = blockIdx.x * 256 + threadIdx.x;
    int e    = gid >> 5;
    int lane = gid & 31;
    if (e >= E) return;
    int s = srcI[e], d = dstI[e];
    int hd = lane >> 3;
    float alpha = exbuf[(size_t)e * NHEAD + hd] / (denom[d * NHEAD + hd] + 1e-16f);
    const float* hp = h + (size_t)s * FDIM + lane * 8;
    float*       ap = agg + (size_t)d * FDIM + lane * 8;
    float4 v0 = *(const float4*)hp;
    float4 v1 = *(const float4*)(hp + 4);
    atomicAdd(ap + 0, v0.x * alpha); atomicAdd(ap + 1, v0.y * alpha);
    atomicAdd(ap + 2, v0.z * alpha); atomicAdd(ap + 3, v0.w * alpha);
    atomicAdd(ap + 4, v1.x * alpha); atomicAdd(ap + 5, v1.y * alpha);
    atomicAdd(ap + 6, v1.z * alpha); atomicAdd(ap + 7, v1.w * alpha);
}

// mode 0: out = relu(agg+b) + res ; mode 1: out = relu(agg+b) ; mode 2: out = relu(agg+b+res)
__global__ __launch_bounds__(256) void finalize_k(const float* __restrict__ agg,
                                                  const float* __restrict__ bias,
                                                  const float* __restrict__ res,
                                                  float* __restrict__ outp,
                                                  size_t total, int mode) {
    size_t i = (size_t)blockIdx.x * 256 + threadIdx.x;
    if (i >= total) return;
    float v = agg[i] + bias[i & 255];
    if (mode == 2) v += res[i];
    v = v > 0.f ? v : 0.f;
    if (mode == 0) v += res[i];
    outp[i] = v;
}

// ---------------- launcher ----------------
extern "C" void kernel_launch(void* const* d_in, const int* in_sizes, int n_in,
                              void* d_out, int out_size, void* d_ws, size_t ws_size,
                              hipStream_t stream) {
    const float* x   = (const float*)d_in[0];
    const int*   ei  = (const int*)d_in[1];
    const float* Wl[3]  = {(const float*)d_in[2], (const float*)d_in[6],  (const float*)d_in[10]};
    const float* asl[3] = {(const float*)d_in[3], (const float*)d_in[7],  (const float*)d_in[11]};
    const float* adl[3] = {(const float*)d_in[4], (const float*)d_in[8],  (const float*)d_in[12]};
    const float* bl[3]  = {(const float*)d_in[5], (const float*)d_in[9],  (const float*)d_in[13]};

    const int N = in_sizes[0] / FDIM;   // 50000
    const int E = in_sizes[1] / 2;      // 500000
    const int* srcI = ei;
    const int* dstI = ei + E;
    const size_t nf = (size_t)N * FDIM;

    char* ws = (char*)d_ws;
    size_t off = 0;
    auto carve = [&](size_t bytes) -> void* {
        void* p = ws + off;
        off = (off + bytes + 255) & ~(size_t)255;
        return p;
    };
    float*    hbuf  = (float*)carve(nf * 4);
    float*    agg   = (float*)carve(nf * 4);
    float*    cur   = (float*)carve(nf * 4);
    __bf16*   xb    = (__bf16*)carve(nf * 2);
    __bf16*   Wb    = (__bf16*)carve((size_t)FDIM * FDIM * 2);
    float*    e_s   = (float*)carve((size_t)N * NHEAD * 4);
    float*    e_d   = (float*)carve((size_t)N * NHEAD * 4);
    unsigned* mato  = (unsigned*)carve((size_t)N * NHEAD * 4);
    float*    denom = (float*)carve((size_t)N * NHEAD * 4);
    float*    exbuf = (float*)carve((size_t)(E + N) * NHEAD * 4);

    const int smemBytes = 128 * FDIM * 2 + FDIM * FDIM * 2;  // 196608

    auto run_layer = [&](const float* feat, const float* W, const float* a_s,
                         const float* a_d, const float* bias, const float* res,
                         float* outp, int mode) {
        cvt_bf16_k<<<(int)((nf + 255) / 256), 256, 0, stream>>>(feat, xb, nf);
        cvt_w_transpose_k<<<(FDIM * FDIM) / 256, 256, 0, stream>>>(W, Wb);
        gemm_wmma_k<<<(N + 127) / 128, 256, smemBytes, stream>>>(xb, Wb, hbuf, N);
        att_k<<<(N * NHEAD + 255) / 256, 256, 0, stream>>>(hbuf, a_s, a_d, e_s, e_d, N);
        init_max_k<<<(N * NHEAD + 255) / 256, 256, 0, stream>>>(e_s, e_d, mato, denom, N);
        edge_max_k<<<(E * NHEAD + 255) / 256, 256, 0, stream>>>(srcI, dstI, e_s, e_d, mato, E);
        ex_denom_k<<<(int)(((size_t)(E + N) * NHEAD + 255) / 256), 256, 0, stream>>>(
            srcI, dstI, e_s, e_d, mato, exbuf, denom, E, N);
        agg_self_k<<<(int)((nf + 255) / 256), 256, 0, stream>>>(hbuf, exbuf, denom, agg, N, E);
        agg_edge_k<<<(E + 7) / 8, 256, 0, stream>>>(hbuf, srcI, dstI, exbuf, denom, agg, E);
        finalize_k<<<(int)((nf + 255) / 256), 256, 0, stream>>>(agg, bias, res, outp, nf, mode);
    };

    run_layer(x,   Wl[0], asl[0], adl[0], bl[0], x,       cur,            0);
    run_layer(cur, Wl[1], asl[1], adl[1], bl[1], nullptr, cur,            1);
    run_layer(cur, Wl[2], asl[2], adl[2], bl[2], x,       (float*)d_out,  2);
}
